// Graph_layer_89197880803734
// MI455X (gfx1250) — matrix operand order repaired and checked by reading, not compiled, not run on hardware
//
#include <hip/hip_runtime.h>

typedef __bf16 bf16;
typedef __attribute__((ext_vector_type(8)))  __bf16 bf16x8;
typedef __attribute__((ext_vector_type(16))) __bf16 v16bf;
typedef __attribute__((ext_vector_type(8)))  float  v8f;

#define BB   4096
#define NSEQ 18
#define NU   16
#define DIN  256
#define HID  128
#define DOUT 256

#define BM 128
#define BN 128
#define BK 32

#ifndef __has_builtin
#define __has_builtin(x) 0
#endif

#if __has_builtin(__builtin_amdgcn_global_load_async_to_lds_b128)
#define HAVE_ASYNC_LDS 1
#else
#define HAVE_ASYNC_LDS 0
#endif

#if __has_builtin(__builtin_amdgcn_s_wait_asynccnt)
#define WAIT_ASYNC(n) __builtin_amdgcn_s_wait_asynccnt(n)
#else
#define WAIT_ASYNC(n) asm volatile("s_wait_asynccnt %0" ::"i"(n))
#endif

__device__ __forceinline__ bf16 f2bf(float f) {
  union { float f; unsigned u; } x; x.f = f;
  unsigned r = (x.u + 0x7FFFu + ((x.u >> 16) & 1u)) >> 16;
  union { unsigned short s; bf16 b; } y; y.s = (unsigned short)r;
  return y.b;
}

#if HAVE_ASYNC_LDS
// b128 async-copy operand type: int32x4 (clang: vector_size(16) int), with
// global (addrspace(1), printed as __device__) / LDS (addrspace(3)) pointers.
typedef int v4i_ __attribute__((vector_size(4 * sizeof(int))));
typedef __attribute__((address_space(1))) v4i_ gv4i;
typedef __attribute__((address_space(3))) v4i_ lv4i;

__device__ __forceinline__ gv4i* to_glb(const void* p) {
  return (gv4i*)(unsigned long long)p;
}
__device__ __forceinline__ lv4i* to_lds(void* p) {
  // generic LDS pointer -> LDS offset (addr[31:0]) -> addrspace(3)
  return (lv4i*)(unsigned)(unsigned long long)p;
}
#endif

// ---------------------------------------------------------------------------
// One K-step of the wave-level MMA: 2x4 tiles of v_wmma_f32_16x16x32_bf16.
// As buffer: BM x BK in natural k order, so each A fragment is a
// contiguous 32B run.  Bs buffer: BN x BK (W pre-transposed), fragments are
// naturally contiguous.
// ---------------------------------------------------------------------------
__device__ __forceinline__ void mma_step(const bf16* __restrict__ AsB,
                                         const bf16* __restrict__ BsB,
                                         int wm, int wn, int lane,
                                         v8f (&acc)[2][4])
{
  const int arow  = lane & 15;
  const int asel  = lane >> 4;         // 0 -> K{0..15}, 1 -> K{16..31}
  const int bcol  = lane & 15;
  const int khalf = (lane >> 4) << 4;  // 0 or 16

  union { bf16x8 h[2]; v16bf v; } af[2];
#pragma unroll
  for (int i = 0; i < 2; ++i) {
    const bf16* p = &AsB[(wm * 32 + i * 16 + arow) * BK + asel * 16];
    af[i].h[0] = *(const bf16x8*)p;
    af[i].h[1] = *(const bf16x8*)(p + 8);
  }
  v16bf bfr[4];
#pragma unroll
  for (int j = 0; j < 4; ++j)
    bfr[j] = *(const v16bf*)&BsB[(wn * 64 + j * 16 + bcol) * BK + khalf];

#pragma unroll
  for (int i = 0; i < 2; ++i)
#pragma unroll
    for (int j = 0; j < 4; ++j)
      acc[i][j] = __builtin_amdgcn_wmma_f32_16x16x32_bf16(
          false, af[i].v, false, bfr[j], (short)0, acc[i][j], false, false);
}

// ---------------------------------------------------------------------------
// Batched bf16 GEMM: C[z] (M x N, f32) = A[z] (M x K bf16) @ W, W given as
// WT = N x K row-major (shared).  Block tile 128x128, K-step 32, 8 waves.
// Async path: triple-buffered LDS fed by GLOBAL_LOAD_ASYNC_TO_LDS_B128 with
// 2 tiles in flight (s_wait_asynccnt 4 + one barrier per K-step).
// Fallback: register-prefetch double-buffered pipeline.
// ---------------------------------------------------------------------------
__global__ __launch_bounds__(256) void gemm_bf16_nt(
    const bf16* __restrict__ A, int lda, long strideA,
    const bf16* __restrict__ WT,
    float* __restrict__ C, int ldc, long strideC, int K)
{
#if HAVE_ASYNC_LDS
  __shared__ __align__(128) bf16 As[3][BM * BK];
  __shared__ __align__(128) bf16 Bs[3][BN * BK];
#else
  __shared__ __align__(128) bf16 As[2][BM * BK];
  __shared__ __align__(128) bf16 Bs[2][BN * BK];
#endif

  const int tid  = threadIdx.x;
  const int lane = tid & 31;
  const int wave = tid >> 5;
  const int wm   = wave >> 1;          // 0..3 -> 32-row slab
  const int wn   = wave & 1;           // 0..1 -> 64-col slab
  const int m0   = blockIdx.y * BM;
  const int n0   = blockIdx.x * BN;

  const bf16* Ab = A + (long)blockIdx.z * strideA;
  float*      Cb = C + (long)blockIdx.z * strideC;

  const v8f vzero = {0.f,0.f,0.f,0.f,0.f,0.f,0.f,0.f};
  v8f acc[2][4];
#pragma unroll
  for (int i = 0; i < 2; ++i)
#pragma unroll
    for (int j = 0; j < 4; ++j) acc[i][j] = vzero;

  // Per-thread staging coordinates: 2 chunks of 8 bf16 (16B) for A and for B.
  const bf16* gA[2]; const bf16* gB[2];
  int ldsAoff[2], ldsBoff[2];
#pragma unroll
  for (int c = 0; c < 2; ++c) {
    int lin = tid + c * 256;
    int r   = lin >> 2;
    int q   = lin & 3;
    int pos = q;                                // identity k-chunk order
    gA[c]      = Ab + (long)(m0 + r) * lda + q * 8;
    ldsAoff[c] = r * BK + pos * 8;
    gB[c]      = WT + (long)(n0 + r) * K + q * 8;
    ldsBoff[c] = r * BK + q * 8;
  }

  const int ntiles = K / BK;

#if HAVE_ASYNC_LDS
  auto issue = [&](int t, int b) {
    long k0 = (long)t * BK;
#pragma unroll
    for (int c = 0; c < 2; ++c) {
      __builtin_amdgcn_global_load_async_to_lds_b128(
          to_glb(gA[c] + k0), to_lds(&As[b][ldsAoff[c]]), 0, 0);
      __builtin_amdgcn_global_load_async_to_lds_b128(
          to_glb(gB[c] + k0), to_lds(&Bs[b][ldsBoff[c]]), 0, 0);
    }
  };
  issue(0, 0);
  if (ntiles > 1) issue(1, 1);
  for (int j = 0; j < ntiles; ++j) {
    // in-order async completion: <=4 outstanding (tile j+1) => tile j resident
    if (j + 1 < ntiles) WAIT_ASYNC(4);
    else                WAIT_ASYNC(0);
    __syncthreads();                 // all waves' tile-j portions visible; also
                                     // certifies compute j-1 done -> buf (j+2)%3 free
    if (j + 2 < ntiles) issue(j + 2, (j + 2) % 3);
    mma_step(As[j % 3], Bs[j % 3], wm, wn, lane, acc);
  }
#else
  bf16x8 ra[2], rb[2];
  auto loadRegs = [&](int t) {
    long k0 = (long)t * BK;
#pragma unroll
    for (int c = 0; c < 2; ++c) {
      ra[c] = *(const bf16x8*)(gA[c] + k0);
      rb[c] = *(const bf16x8*)(gB[c] + k0);
    }
  };
  auto storeLds = [&](int b) {
#pragma unroll
    for (int c = 0; c < 2; ++c) {
      *(bf16x8*)&As[b][ldsAoff[c]] = ra[c];
      *(bf16x8*)&Bs[b][ldsBoff[c]] = rb[c];
    }
  };
  loadRegs(0);
  storeLds(0);
  __syncthreads();
  for (int j = 0; j < ntiles; ++j) {
    if (j + 1 < ntiles) loadRegs(j + 1);   // global latency hidden by WMMAs
    mma_step(As[j & 1], Bs[j & 1], wm, wn, lane, acc);
    if (j + 1 < ntiles) storeLds((j + 1) & 1);
    __syncthreads();
  }
#endif

  // C/D layout: VGPR r -> row r (lanes 0-15) / row r+8 (lanes 16-31), col = lane%16
  const int rbase = m0 + wm * 32 + (lane >> 4) * 8;
  const int cc    = n0 + wn * 64 + (lane & 15);
#pragma unroll
  for (int i = 0; i < 2; ++i)
#pragma unroll
    for (int j = 0; j < 4; ++j)
#pragma unroll
      for (int r = 0; r < 8; ++r)
        Cb[(long)(rbase + i * 16 + r) * ldc + cc + j * 16] = acc[i][j][r];
}

// ---------------------------------------------------------------------------
// Deterministic batch-norm statistics: two-stage column sum / sum-of-squares.
// ---------------------------------------------------------------------------
__global__ __launch_bounds__(256) void col_stats_partial(
    const float* __restrict__ C, int ncols, long strideC, int rowsPerBlock,
    float* __restrict__ partial, int nRB)
{
  const int c  = threadIdx.x;                 // blockDim.x == ncols
  const int rb = blockIdx.x;
  const int bz = blockIdx.y;
  const float* Cb = C + (long)bz * strideC + (long)rb * rowsPerBlock * ncols;
  float s = 0.f, q = 0.f;
  for (int r = 0; r < rowsPerBlock; ++r) {
    float v = Cb[(long)r * ncols + c];
    s += v; q += v * v;
  }
  float* P = partial + ((long)bz * nRB + rb) * 2 * ncols;
  P[c] = s;
  P[ncols + c] = q;
}

__global__ __launch_bounds__(256) void col_stats_final(
    const float* __restrict__ partial, int ncols, int nRB, int M,
    const float* __restrict__ g, const float* __restrict__ be,
    float* __restrict__ ss)
{
  const int c  = threadIdx.x;
  const int bz = blockIdx.x;
  const float* P = partial + (long)bz * nRB * 2 * ncols;
  float s = 0.f, q = 0.f;
  for (int rb = 0; rb < nRB; ++rb) {
    s += P[(long)rb * 2 * ncols + c];
    q += P[(long)rb * 2 * ncols + ncols + c];
  }
  const float invM = 1.0f / (float)M;
  float mu  = s * invM;
  float var = q * invM - mu * mu;
  float scale = g[c] * rsqrtf(var + 1e-5f);
  float* S = ss + (long)bz * 2 * ncols;
  S[c] = scale;
  S[ncols + c] = be[c] - mu * scale;   // pre-BN bias cancels in (h - mean)
}

// ---------------------------------------------------------------------------
// y = sigmoid(scale*c + shift), written strided.  ncols = 1 << cshift.
// ---------------------------------------------------------------------------
__global__ __launch_bounds__(256) void bn_sigmoid_f32(
    const float* __restrict__ C, int cshift, long strideC,
    const float* __restrict__ ss,
    float* __restrict__ out, long ldo, long strideOut, int coloff)
{
  const int bz = blockIdx.y;
  const int ncols = 1 << cshift;
  long e = (long)blockIdx.x * 256 + threadIdx.x;
  int row = (int)(e >> cshift);
  int col = (int)(e & (ncols - 1));
  const float* S = ss + (long)bz * 2 * ncols;
  float v = C[(long)bz * strideC + e];
  float y = v * S[col] + S[ncols + col];
  y = 1.0f / (1.0f + expf(-y));
  out[(long)bz * strideOut + (long)row * ldo + coloff + col] = y;
}

__global__ __launch_bounds__(256) void bn_sigmoid_bf16(
    const float* __restrict__ C, int cshift, long strideC,
    const float* __restrict__ ss,
    bf16* __restrict__ out, long ldo, long strideOut, int coloff)
{
  const int bz = blockIdx.y;
  const int ncols = 1 << cshift;
  long e = (long)blockIdx.x * 256 + threadIdx.x;
  int row = (int)(e >> cshift);
  int col = (int)(e & (ncols - 1));
  const float* S = ss + (long)bz * 2 * ncols;
  float v = C[(long)bz * strideC + e];
  float y = v * S[col] + S[ncols + col];
  y = 1.0f / (1.0f + expf(-y));
  out[(long)bz * strideOut + (long)row * ldo + coloff + col] = f2bf(y);
}

// ---------------------------------------------------------------------------
// Conversions and glue
// ---------------------------------------------------------------------------
__global__ __launch_bounds__(256) void f32_to_bf16_k(
    const float* __restrict__ src, bf16* __restrict__ dst, long n)
{
  long i = (long)blockIdx.x * 256 + threadIdx.x;
  if (i < n) dst[i] = f2bf(src[i]);
}

// W (K x N f32, row-major) -> WT (N x K bf16, row-major)
__global__ __launch_bounds__(256) void transpose_w(
    const float* __restrict__ w, bf16* __restrict__ wt, int K, int N)
{
  long i = (long)blockIdx.x * 256 + threadIdx.x;
  int n = (int)(i / K);
  int k = (int)(i - (long)n * K);
  wt[i] = f2bf(w[(long)k * N + n]);
}

// top-2 / leave-one-out across 16 users; writes LOO into comb col block 0
__global__ __launch_bounds__(256) void topk_loo(
    const float* __restrict__ U, bf16* __restrict__ comb)
{
  long idx = (long)blockIdx.x * 256 + threadIdx.x;   // b*256 + d
  int b = (int)(idx >> 8);
  int d = (int)(idx & 255);
  float m1 = -3.0e38f, m2 = -3.0e38f; int a1 = 0;
  for (int n = 0; n < NU; ++n) {
    float v = U[(long)n * BB * DOUT + idx];
    if (v > m1) { m2 = m1; m1 = v; a1 = n; }
    else if (v > m2) { m2 = v; }
  }
  for (int n = 0; n < NU; ++n)
    comb[((long)n * BB + b) * 1024 + d] = f2bf(n == a1 ? m2 : m1);
}

// mean over 16 user nets -> bf16 into comb768 cols 0..255
__global__ __launch_bounds__(256) void mean16(
    const float* __restrict__ U, bf16* __restrict__ out, int ldo)
{
  long idx = (long)blockIdx.x * 256 + threadIdx.x;
  int b = (int)(idx >> 8);
  int c = (int)(idx & 255);
  float s = 0.f;
  for (int n = 0; n < NU; ++n) s += U[(long)n * BB * DOUT + idx];
  out[(long)b * ldo + c] = f2bf(s * 0.0625f);
}

// fill comb cols 256..1023: [irs_f | bs_f | xu_n]
__global__ __launch_bounds__(256) void assemble_comb(
    const bf16* __restrict__ irsf, const bf16* __restrict__ bsf,
    const bf16* __restrict__ xbf, bf16* __restrict__ comb)
{
  int b = blockIdx.x;
  int n = blockIdx.y;
  int c = threadIdx.x;    // 256
  bf16* row = comb + ((long)n * BB + b) * 1024;
  row[256 + c] = irsf[(long)b * DOUT + c];
  row[512 + c] = bsf[(long)b * DOUT + c];
  row[768 + c] = xbf[(long)b * (NSEQ * DIN) + n * DIN + c];
}

__global__ __launch_bounds__(256) void copy_slice(
    const bf16* __restrict__ src, long lds_, bf16* __restrict__ dst,
    long ldd, int coloff)
{
  int b = blockIdx.x;
  int c = threadIdx.x;
  dst[(long)b * ldd + coloff + c] = src[(long)b * lds_ + c];
}

// ---------------------------------------------------------------------------
// Host-side orchestration
// ---------------------------------------------------------------------------
struct NetDev {
  const float *g1, *be1, *g2, *be2;
  const bf16 *w1t, *w2t;
  int din;
};
struct Scratch { float* C1; float* C2; bf16* A1; float* partial; float* ss; };

static void run_net(hipStream_t s, const NetDev& net,
                    const bf16* A, int lda, long strideA, int nb,
                    const Scratch& sc,
                    void* out, long ldo, long strideOut, int coloff, bool outBF16)
{
  const long sC1 = (long)BB * HID;
  const long sC2 = (long)BB * DOUT;
  // layer 1: (B x din) @ (din x 128)
  gemm_bf16_nt<<<dim3(HID / BN, BB / BM, nb), 256, 0, s>>>(
      A, lda, strideA, net.w1t, sc.C1, HID, sC1, net.din);
  col_stats_partial<<<dim3(32, nb), HID, 0, s>>>(sc.C1, HID, sC1, BB / 32, sc.partial, 32);
  col_stats_final<<<nb, HID, 0, s>>>(sc.partial, HID, 32, BB, net.g1, net.be1, sc.ss);
  bn_sigmoid_bf16<<<dim3((BB * HID) / 256, nb), 256, 0, s>>>(
      sc.C1, 7, sC1, sc.ss, sc.A1, HID, sC1, 0);
  // layer 2: (B x 128) @ (128 x 256)
  gemm_bf16_nt<<<dim3(DOUT / BN, BB / BM, nb), 256, 0, s>>>(
      sc.A1, HID, sC1, net.w2t, sc.C2, DOUT, sC2, HID);
  col_stats_partial<<<dim3(32, nb), DOUT, 0, s>>>(sc.C2, DOUT, sC2, BB / 32, sc.partial, 32);
  col_stats_final<<<nb, DOUT, 0, s>>>(sc.partial, DOUT, 32, BB, net.g2, net.be2, sc.ss);
  if (outBF16)
    bn_sigmoid_bf16<<<dim3((BB * DOUT) / 256, nb), 256, 0, s>>>(
        sc.C2, 8, sC2, sc.ss, (bf16*)out, ldo, strideOut, coloff);
  else
    bn_sigmoid_f32<<<dim3((BB * DOUT) / 256, nb), 256, 0, s>>>(
        sc.C2, 8, sC2, sc.ss, (float*)out, ldo, strideOut, coloff);
}

extern "C" void kernel_launch(void* const* d_in, const int* in_sizes, int n_in,
                              void* d_out, int out_size, void* d_ws, size_t ws_size,
                              hipStream_t stream)
{
  if (n_in < 81) return;

  // Input order: either JAX-sorted-keys pytree (params first, x last) or
  // insertion order (x first). Detect via x's element count.
  const bool x_first = (in_sizes[0] == BB * NSEQ * DIN);
  const int  pbase   = x_first ? 1 : 0;
  const float* x = (const float*)d_in[x_first ? 0 : 80];

  // roles: 0 pu.user 1 pu.irs 2 pu.bs 3 pu.comb 4 pi.ue 5 pi.bs 6 pi.comb
  //        7 pb.ue 8 pb.irs 9 pb.comb
  const int din_role[10]   = {256, 256, 256, 1024, 256, 256, 768, 256, 256, 768};
  const int pos_sorted[10] = {9, 8, 6, 7, 5, 3, 4, 2, 1, 0};   // keys sorted
  const int pos_insert[10] = {0, 1, 2, 3, 4, 5, 6, 7, 8, 9};   // insertion
  // field index within net for: g1, be1, g2, be2, w1, w2
  const int fld_sorted[6] = {4, 2, 5, 3, 6, 7};  // b1,b2,be1,be2,g1,g2,w1,w2
  const int fld_insert[6] = {2, 3, 6, 7, 0, 4};  // w1,b1,g1,be1,w2,b2,g2,be2
  const int* pos = x_first ? pos_insert : pos_sorted;
  const int* fld = x_first ? fld_insert : fld_sorted;

  // ---- workspace carve -------------------------------------------------
  char* base = (char*)d_ws;
  size_t off = 0;
  auto alloc = [&](size_t bytes) -> void* {
    void* p = base + off;
    off = (off + bytes + 255) & ~(size_t)255;
    return p;
  };

  bf16* xbf = (bf16*)alloc((size_t)BB * NSEQ * DIN * sizeof(bf16));

  NetDev nets[10];
  const float* w1src[10]; const float* w2src[10];
  bf16* w1dst[10]; bf16* w2dst[10];
  for (int r = 0; r < 10; ++r) {
    int bi = pbase + 8 * pos[r];
    nets[r].g1  = (const float*)d_in[bi + fld[0]];
    nets[r].be1 = (const float*)d_in[bi + fld[1]];
    nets[r].g2  = (const float*)d_in[bi + fld[2]];
    nets[r].be2 = (const float*)d_in[bi + fld[3]];
    w1src[r] = (const float*)d_in[bi + fld[4]];
    w2src[r] = (const float*)d_in[bi + fld[5]];
    nets[r].din = din_role[r];
    w1dst[r] = (bf16*)alloc((size_t)HID * din_role[r] * sizeof(bf16));
    w2dst[r] = (bf16*)alloc((size_t)DOUT * HID * sizeof(bf16));
    nets[r].w1t = w1dst[r];
    nets[r].w2t = w2dst[r];
  }

  Scratch sc;
  sc.C1      = (float*)alloc((size_t)NU * BB * HID * sizeof(float));
  sc.C2      = (float*)alloc((size_t)NU * BB * DOUT * sizeof(float));
  sc.A1      = (bf16*)alloc((size_t)NU * BB * HID * sizeof(bf16));
  sc.partial = (float*)alloc((size_t)NU * 32 * 2 * DOUT * sizeof(float));
  sc.ss      = (float*)alloc((size_t)NU * 2 * DOUT * sizeof(float));
  bf16* comb  = (bf16*)alloc((size_t)NU * BB * 1024 * sizeof(bf16));
  bf16* comb3 = (bf16*)alloc((size_t)BB * 768 * sizeof(bf16));
  bf16* irsf  = (bf16*)alloc((size_t)BB * DOUT * sizeof(bf16));
  bf16* bsf   = (bf16*)alloc((size_t)BB * DOUT * sizeof(bf16));
  if (off > ws_size) return;   // workspace too small; nothing safe to do

  const int ldx = NSEQ * DIN;  // 4608

  // ---- one-time conversions -------------------------------------------
  f32_to_bf16_k<<<(BB * NSEQ * DIN) / 256, 256, 0, stream>>>(
      x, xbf, (long)BB * NSEQ * DIN);
  for (int r = 0; r < 10; ++r) {
    transpose_w<<<(din_role[r] * HID) / 256, 256, 0, stream>>>(
        w1src[r], w1dst[r], din_role[r], HID);
    transpose_w<<<(HID * DOUT) / 256, 256, 0, stream>>>(
        w2src[r], w2dst[r], HID, DOUT);
  }

  // ---- user branch -----------------------------------------------------
  // u = pu.user(xu) for all 16 users (batched, f32 in-place in C2)
  run_net(stream, nets[0], xbf, ldx, DIN, NU, sc,
          sc.C2, DOUT, (long)BB * DOUT, 0, false);
  topk_loo<<<(BB * DOUT) / 256, 256, 0, stream>>>(sc.C2, comb);
  run_net(stream, nets[1], xbf + NU * DIN, ldx, 0, 1, sc,
          irsf, DOUT, 0, 0, true);                      // pu.irs(x_irs)
  run_net(stream, nets[2], xbf + (NU + 1) * DIN, ldx, 0, 1, sc,
          bsf, DOUT, 0, 0, true);                       // pu.bs(x_bs)
  assemble_comb<<<dim3(BB, NU), 256, 0, stream>>>(irsf, bsf, xbf, comb);
  run_net(stream, nets[3], comb, 1024, (long)BB * 1024, NU, sc,
          d_out, ldx, DOUT, 0, false);                  // user_out -> out[:, 0:16, :]

  // ---- irs branch ------------------------------------------------------
  run_net(stream, nets[4], xbf, ldx, DIN, NU, sc,
          sc.C2, DOUT, (long)BB * DOUT, 0, false);      // pi.ue batched
  mean16<<<(BB * DOUT) / 256, 256, 0, stream>>>(sc.C2, comb3, 768);
  run_net(stream, nets[5], xbf + (NU + 1) * DIN, ldx, 0, 1, sc,
          comb3, 768, 0, 256, true);                    // pi.bs(x_bs) -> cols 256..511
  copy_slice<<<BB, 256, 0, stream>>>(xbf + NU * DIN, ldx, comb3, 768, 512);
  run_net(stream, nets[6], comb3, 768, 0, 1, sc,
          d_out, ldx, 0, NU * DOUT, false);             // irs_out -> out[:, 16, :]

  // ---- bs branch -------------------------------------------------------
  run_net(stream, nets[7], xbf, ldx, DIN, NU, sc,
          sc.C2, DOUT, (long)BB * DOUT, 0, false);      // pb.ue batched
  mean16<<<(BB * DOUT) / 256, 256, 0, stream>>>(sc.C2, comb3, 768);
  run_net(stream, nets[8], xbf + NU * DIN, ldx, 0, 1, sc,
          comb3, 768, 0, 256, true);                    // pb.irs(x_irs)
  copy_slice<<<BB, 256, 0, stream>>>(xbf + (NU + 1) * DIN, ldx, comb3, 768, 512);
  run_net(stream, nets[9], comb3, 768, 0, 1, sc,
          d_out, ldx, 0, (NU + 1) * DOUT, false);       // bs_out -> out[:, 17, :]

  (void)out_size;
}
